// GCN_80616536146364
// MI455X (gfx1250) — matrix operand order repaired and checked by reading, not compiled, not run on hardware
//
#include <hip/hip_runtime.h>
#include <hip/hip_bf16.h>

// ---------------------------------------------------------------------------
// GCN forward for MI455X (gfx1250): bf16 WMMA GEMMs + f32 atomic scatter.
// A-tile staging uses gfx1250 async global->LDS copies when available.
// ---------------------------------------------------------------------------

typedef __attribute__((ext_vector_type(16))) __bf16 v16bf;
typedef __attribute__((ext_vector_type(8)))  __bf16 v8bf;
typedef __attribute__((ext_vector_type(8)))  float  v8f;

#define BM 128
#define BN 64
#define BK 32

#if defined(__has_builtin)
#  if __has_builtin(__builtin_amdgcn_global_load_async_to_lds_b128)
#    define USE_ASYNC_LDS 1
#  endif
#  if __has_builtin(__builtin_amdgcn_s_wait_asynccnt)
#    define HAVE_WAIT_ASYNC 1
#  endif
#endif

#ifdef USE_ASYNC_LDS
// Exact parameter types per hipcc diagnostic:
//   param0: 'int __vector(4) __device__ *'  (AS1 global)
//   param1: LDS destination (AS3)
typedef int v4i __attribute__((vector_size(4 * sizeof(int))));
typedef __attribute__((address_space(1))) v4i as1_v4i;
typedef __attribute__((address_space(3))) v4i as3_v4i;
#endif

// ---------------- f32 -> bf16 conversion ----------------
__global__ __launch_bounds__(256)
void f32_to_bf16_kernel(const float* __restrict__ in, __bf16* __restrict__ out,
                        long long n) {
  long long i = (long long)blockIdx.x * blockDim.x + threadIdx.x;
  if (i < n) out[i] = (__bf16)in[i];
}

// ---------------- edge-weighted scatter-add (bf16 msgs -> f32 atomics) ------
__global__ __launch_bounds__(256)
void scatter_kernel(const __bf16* __restrict__ h, const int* __restrict__ ei,
                    const float* __restrict__ ew, float* __restrict__ agg,
                    int E, int C) {
  long long idx = (long long)blockIdx.x * blockDim.x + threadIdx.x;
  int perEdge = C >> 2;                       // 4 channels per thread
  long long total = (long long)E * perEdge;
  if (idx >= total) return;
  int e  = (int)(idx / perEdge);
  int c0 = ((int)(idx % perEdge)) << 2;
  int src = ei[e];
  int dst = ei[E + e];
  float w = ew[e];
  const __bf16* hp = h + (long long)src * C + c0;
  float* ap = agg + (long long)dst * C + c0;
  atomicAdd(ap + 0, w * (float)hp[0]);
  atomicAdd(ap + 1, w * (float)hp[1]);
  atomicAdd(ap + 2, w * (float)hp[2]);
  atomicAdd(ap + 3, w * (float)hp[3]);
}

// ---------------- bias + PReLU, dual-format store ----------------
__global__ __launch_bounds__(256)
void bias_prelu_kernel(const float* __restrict__ agg, const float* __restrict__ bias,
                       const float* __restrict__ alpha, __bf16* __restrict__ out_bf,
                       float* __restrict__ out_f32, long long total, int C) {
  long long i = (long long)blockIdx.x * blockDim.x + threadIdx.x;
  if (i >= total) return;
  int c = (int)(i % C);
  float v = agg[i] + bias[c];
  float al = alpha[0];
  v = (v >= 0.0f) ? v : al * v;
  out_bf[i] = (__bf16)v;
  if (out_f32) out_f32[i] = v;
}

// ---------------- tiled bf16 WMMA GEMM: C[M,N] = A[M,K] @ B[K,N] ------------
// Block: 256 threads = 8 waves; tile 128x64, K-step 32.
// Wave w computes rows [w*16, w*16+16) x all 4 16-col sub-tiles.
__global__ __launch_bounds__(256)
void gemm_bf16_wmma_kernel(const __bf16* __restrict__ A, const __bf16* __restrict__ B,
                           int M, int N, int K, int lda, int ldb,
                           void* __restrict__ Cout, int ldc, int storeBf16,
                           const float* __restrict__ alpha) {
  __shared__ alignas(32) __bf16 As[BM * BK];   // row-major [128][32]
  __shared__ alignas(32) __bf16 Bt[BN * BK];   // transposed [64][32] (n-major)

  const int tid  = threadIdx.x;
  const int wave = tid >> 5;
  const int lane = tid & 31;
  const int l16  = lane & 15;
  const int hi   = lane >> 4;     // lane half selects contiguous K half (k = 16*hi + s)

  const int mBlock = blockIdx.x * BM;
  const int nBlock = blockIdx.y * BN;

  v8f acc[4];
#pragma unroll
  for (int j = 0; j < 4; ++j)
#pragma unroll
    for (int i = 0; i < 8; ++i) acc[j][i] = 0.0f;

  const int br = tid >> 3;            // B tile k-row   (0..31)
  const int bc = (tid & 7) * 8;       // B tile col chunk (0..56)

  for (int k0 = 0; k0 < K; k0 += BK) {
    // ---- A tile: 128x32 bf16 = 512 x 16B chunks; 2 chunks per thread.
#ifdef USE_ASYNC_LDS
    {
#pragma unroll
      for (int pass = 0; pass < 2; ++pass) {
        int c   = pass * 256 + tid;       // chunk id (0..511)
        int r   = c >> 2;                 // tile row  (0..127)
        int col = (c & 3) * 8;            // col chunk (0/8/16/24)
        int gr  = mBlock + r;
        if (gr < M) {                     // EXEC-predicated async copy; rows >= M
          const __bf16* gp = A + (long long)gr * lda + k0 + col;  // feed only
          __builtin_amdgcn_global_load_async_to_lds_b128(         // discarded C rows
              (as1_v4i*)gp, (as3_v4i*)&As[r * BK + col], 0, 0);
        }
      }
    }
#else
    {
      int r  = tid >> 1;
      int ah = (tid & 1) * 16;
      int gr = mBlock + r;
      v8bf lo, hi8;
      if (gr < M) {
        const __bf16* gp = A + (long long)gr * lda + k0 + ah;
        lo  = *reinterpret_cast<const v8bf*>(gp);
        hi8 = *reinterpret_cast<const v8bf*>(gp + 8);
      } else {
#pragma unroll
        for (int i = 0; i < 8; ++i) { lo[i] = (__bf16)0.0f; hi8[i] = (__bf16)0.0f; }
      }
      *reinterpret_cast<v8bf*>(&As[r * BK + ah])     = lo;
      *reinterpret_cast<v8bf*>(&As[r * BK + ah + 8]) = hi8;
    }
#endif
    // ---- B tile: load row-major, store transposed into LDS
    {
      const __bf16* gp = B + (long long)(k0 + br) * ldb + nBlock + bc;
      v8bf v = *reinterpret_cast<const v8bf*>(gp);
#pragma unroll
      for (int i = 0; i < 8; ++i) Bt[(bc + i) * BK + br] = v[i];
    }
#ifdef USE_ASYNC_LDS
#  ifdef HAVE_WAIT_ASYNC
    __builtin_amdgcn_s_wait_asynccnt(0);
#  else
    asm volatile("s_wait_asynccnt 0x0" ::: "memory");
#  endif
#endif
    __syncthreads();

    // ---- fragments + WMMA
    {
      // A 16x32 bf16 fragment: lanes 0-15 -> K 0-15,
      // lanes 16-31 -> K 16-31; row = lane&15.
      const int arow = wave * 16 + l16;
      const int ab   = hi ? 16 : 0;
      v8bf alo = *reinterpret_cast<const v8bf*>(&As[arow * BK + ab]);
      v8bf ahi = *reinterpret_cast<const v8bf*>(&As[arow * BK + ab + 8]);
      v16bf a;
#pragma unroll
      for (int i = 0; i < 8; ++i) { a[i] = alo[i]; a[i + 8] = ahi[i]; }

      // B 32x16 fragment: col = lane&15, K = 0-15 (lanes 0-15) / 16-31 (16-31);
      // contiguous in transposed LDS tile -> one 32B read.
      const int kb = hi * 16;
#pragma unroll
      for (int j = 0; j < 4; ++j) {
        v16bf b = *reinterpret_cast<const v16bf*>(&Bt[(j * 16 + l16) * BK + kb]);
        acc[j] = __builtin_amdgcn_wmma_f32_16x16x32_bf16(
            /*neg_a=*/false, a, /*neg_b=*/false, b,
            /*c_mod=*/(short)0, acc[j], /*reuse_a=*/false, /*reuse_b=*/false);
      }
    }
    __syncthreads();
  }

  // ---- epilogue: C/D layout: VGPR i -> row (hi?8:0)+i, col = lane&15
  const int rowBase = mBlock + wave * 16 + hi * 8;
  const bool useP = (alpha != nullptr);
  const float al = useP ? alpha[0] : 0.0f;
#pragma unroll
  for (int j = 0; j < 4; ++j) {
    int gc = nBlock + j * 16 + l16;
#pragma unroll
    for (int i = 0; i < 8; ++i) {
      int gr = rowBase + i;
      if (gr < M) {
        float v = acc[j][i];
        if (useP) v = (v >= 0.0f) ? v : al * v;
        if (storeBf16) ((__bf16*)Cout)[(long long)gr * ldc + gc] = (__bf16)v;
        else           ((float*) Cout)[(long long)gr * ldc + gc] = v;
      }
    }
  }
}

// ---------------------------------------------------------------------------
extern "C" void kernel_launch(void* const* d_in, const int* in_sizes, int n_in,
                              void* d_out, int out_size, void* d_ws, size_t ws_size,
                              hipStream_t stream) {
  (void)in_sizes; (void)n_in; (void)out_size; (void)ws_size;
  const int N = 50000, E = 800000, IN = 256, H = 128, OUT = 64;
  const int H2 = 2 * H;      // 256
  const int CAT = 4 * H;     // 512

  const float* seq  = (const float*)d_in[0];
  const int*   ei   = (const int*)  d_in[1];
  const float* ew   = (const float*)d_in[2];
  const float* W1   = (const float*)d_in[3];
  const float* b1   = (const float*)d_in[4];
  const float* W2   = (const float*)d_in[5];
  const float* b2   = (const float*)d_in[6];
  const float* a1   = (const float*)d_in[7];
  const float* a2   = (const float*)d_in[8];
  const float* a3   = (const float*)d_in[9];
  const float* Wfc1 = (const float*)d_in[10];
  const float* Wfc2 = (const float*)d_in[11];
  const float* Wfc3 = (const float*)d_in[12];
  const float* Wfc4 = (const float*)d_in[13];

  char* ws = (char*)d_ws;
  size_t off = 0;
  auto alloc = [&](size_t bytes) -> void* {
    void* p = ws + off;
    off = (off + bytes + 255) & ~(size_t)255;
    return p;
  };

  __bf16* W1b  = (__bf16*)alloc((size_t)IN  * H2  * 2);
  __bf16* W2b  = (__bf16*)alloc((size_t)H2  * H   * 2);
  __bf16* Wf1b = (__bf16*)alloc((size_t)H   * H2  * 2);
  __bf16* Wf2b = (__bf16*)alloc((size_t)IN  * H2  * 2);
  __bf16* Wf3b = (__bf16*)alloc((size_t)CAT * H   * 2);
  __bf16* Wf4b = (__bf16*)alloc((size_t)H   * OUT * 2);
  __bf16* B0   = (__bf16*)alloc((size_t)N * IN  * 2);  // seq bf16, later t bf16
  float*  B2   = (float*) alloc((size_t)N * H2  * 4);  // f32 scatter accumulator
  __bf16* B3   = (__bf16*)alloc((size_t)N * H2  * 2);  // h1 / x1 / x bf16
  __bf16* B1   = (__bf16*)alloc((size_t)N * H   * 2);  // h2 bf16
  __bf16* B4   = (__bf16*)alloc((size_t)N * CAT * 2);  // cat = [s, feat1] bf16
  (void)alloc((size_t)BM * 64 * 2);                    // tail slack for edge tiles

  auto conv = [&](const float* s, __bf16* d, long long n) {
    unsigned blocks = (unsigned)((n + 255) / 256);
    hipLaunchKernelGGL(f32_to_bf16_kernel, dim3(blocks), dim3(256), 0, stream, s, d, n);
  };
  auto gemm = [&](const __bf16* A, const __bf16* Bm, int M, int Nn, int K,
                  int lda, int ldb, void* C, int ldc, int bf16out, const float* alpha) {
    dim3 grid((unsigned)((M + BM - 1) / BM), (unsigned)(Nn / BN));
    hipLaunchKernelGGL(gemm_bf16_wmma_kernel, grid, dim3(256), 0, stream,
                       A, Bm, M, Nn, K, lda, ldb, C, ldc, bf16out, alpha);
  };
  auto scatter = [&](const __bf16* h, float* agg, int C) {
    long long tot = (long long)E * (C / 4);
    hipLaunchKernelGGL(scatter_kernel, dim3((unsigned)((tot + 255) / 256)), dim3(256),
                       0, stream, h, ei, ew, agg, E, C);
  };
  auto biasprelu = [&](const float* agg, const float* bias, const float* alpha,
                       __bf16* obf, float* of32, int C) {
    long long tot = (long long)N * C;
    hipLaunchKernelGGL(bias_prelu_kernel, dim3((unsigned)((tot + 255) / 256)), dim3(256),
                       0, stream, agg, bias, alpha, obf, of32, tot, C);
  };

  float* x_out = (float*)d_out;                  // [N, H]
  float* f_out = (float*)d_out + (size_t)N * H;  // [N, OUT]

  // ---- precision conversion of reused operands
  conv(W1,   W1b,  (long long)IN  * H2);
  conv(W2,   W2b,  (long long)H2  * H);
  conv(Wfc1, Wf1b, (long long)H   * H2);
  conv(Wfc2, Wf2b, (long long)IN  * H2);
  conv(Wfc3, Wf3b, (long long)CAT * H);
  conv(Wfc4, Wf4b, (long long)H   * OUT);
  conv(seq,  B0,   (long long)N   * IN);

  // ---- GCN layer 1: h1 = seq @ W1 ; agg = scatter(w * h1[src]) ; x1 = prelu(agg+b1)
  gemm(B0, W1b, N, H2, IN, IN, H2, B3, H2, /*bf16*/1, nullptr);
  (void)hipMemsetAsync(B2, 0, (size_t)N * H2 * 4, stream);
  scatter(B3, B2, H2);
  biasprelu(B2, b1, a1, B3, nullptr, H2);

  // ---- GCN layer 2: h2 = x1 @ W2 ; agg ; x = prelu(agg+b2) -> d_out (f32) + bf16
  gemm(B3, W2b, N, H, H2, H2, H, B1, H, /*bf16*/1, nullptr);
  (void)hipMemsetAsync(B2, 0, (size_t)N * H * 4, stream);
  scatter(B1, B2, H);
  biasprelu(B2, b2, a2, B3, x_out, H);

  // ---- MLP head: cat = [seq@Wfc2, x@Wfc1] ; feat1 = prelu((cat@Wfc3)@Wfc4, a3)
  gemm(B3, Wf1b, N, H2, H,  H,  H2, B4 + H2, CAT, /*bf16*/1, nullptr); // cols 256..511
  gemm(B0, Wf2b, N, H2, IN, IN, H2, B4,      CAT, /*bf16*/1, nullptr); // cols 0..255
  gemm(B4, Wf3b, N, H,  CAT, CAT, H, B0, H, /*bf16*/1, nullptr);       // t -> B0 (reuse)
  gemm(B0, Wf4b, N, OUT, H, H, OUT, f_out, OUT, /*f32*/0, a3);         // fused PReLU
}
